// GeometricBilinearLayer_55130200212214
// MI455X (gfx1250) — matrix operand order repaired and checked by reading, not compiled, not run on hardware
//
#include <hip/hip_runtime.h>
#include <cstdint>

// ---------------------------------------------------------------------------
// CDNA5 (gfx1250) GeometricBilinearLayer
//   x:[8,2048,64,16] f32, ref:[8,2048,1,16], w_*:[64,64,9]  -> out:[8,2048,128,16]
// Strategy: per-blade channel-mixing GEMMs in bf16 WMMA (16x16x32, f32 acc),
// then fp32 sparse Clifford bilinears (GP: 192 terms, JOIN: 81 terms).
// ---------------------------------------------------------------------------

typedef __attribute__((ext_vector_type(16))) __bf16 v16bf;
typedef __attribute__((ext_vector_type(8)))  float  v8f;

// ----------------------- compile-time Clifford tables ----------------------
constexpr int pcnt(int x){ int c=0; while(x){ c += x&1; x >>= 1; } return c; }
constexpr float csign(int a, int b){ int s=0; a >>= 1; while(a){ s += pcnt(a&b); a >>= 1; } return (s&1) ? -1.0f : 1.0f; }

struct Term { int i, j, k; float s; };
struct GPTab { Term t[192]; };
struct JNTab { Term t[81]; };

constexpr GPTab make_gp(){
  GPTab g{}; int n = 0;
  for (int i=0;i<16;i++) for (int j=0;j<16;j++)
    if (!(i & j & 1)) { g.t[n].i=i; g.t[n].j=j; g.t[n].k=i^j; g.t[n].s=csign(i,j); n++; }
  return g;
}
constexpr JNTab make_jn(){
  JNTab g{}; int n = 0;
  for (int i=0;i<16;i++) for (int j=0;j<16;j++){
    int i15 = i^15, j15 = j^15;
    if ((i15 & j15) == 0){
      int r = i15 | j15, k = r^15;
      g.t[n].i=i; g.t[n].j=j; g.t[n].k=k;
      g.t[n].s = csign(i,i15)*csign(j,j15)*csign(i15,j15)*csign(k,r);
      n++;
    }
  }
  return g;
}
constexpr GPTab GPT = make_gp();
constexpr JNTab JNT = make_jn();

// ----------------------------- bf16 helpers --------------------------------
__device__ __forceinline__ uint16_t f2bf(float f){
  uint32_t u = __float_as_uint(f);
  u += 0x7FFFu + ((u >> 16) & 1u);          // round-to-nearest-even
  return (uint16_t)(u >> 16);
}
__device__ __forceinline__ float bf2f(uint16_t h){
  return __uint_as_float(((uint32_t)h) << 16);
}

union FragB16 { v16bf v; uint4 q[2]; uint16_t h[16]; };

// ----------------------------- geometry constants --------------------------
#define TOK_TILE   16
#define XS_STRIDE  1032                  // uint16 per token (1024 data + 8 pad, 16B aligned, 4-bank skew)
#define GB_BS      24                    // uint16 per (tok,ch) record (16 blades + 8 pad, 48B: 16B aligned)
#define GB_SET     (TOK_TILE*64*GB_BS)   // uint16 per weight-set buffer
#define NPACK      (4*16*4*4*32*8)       // packed-weight dwords: set,blade,ntile,kchunk,lane,vreg

// ---------------------------------------------------------------------------
// Phase 0: pack 4 weight tensors [64(j),64(i),9] f32 into per-blade [K=128,N=64]
// bf16 matrices, pre-swizzled in WMMA B-fragment per-lane order.
//   rows k<64  : coeff of x[.,i=k,  y  ] = w[j,k,   pc(y)]
//   rows k>=64 : coeff of x[.,i=k-64,y^1] = w[j,k-64,4+pc(y)]  (odd y only, else 0)
// B-fragment layout (16-bit, 32x16): lane L holds column n=L&15;
//   k(L, vreg v, chunk c) = c*32 + (L>=16 ? 8 : 0) + (v<4 ? 2v : 16 + 2(v-4))  (pair in one dword)
// ---------------------------------------------------------------------------
__device__ __forceinline__ float wval(const float* w, int y, int k, int n){
  int i = k & 63;
  int b;
  if (k < 64) { b = __popc(y); }
  else { if (!(y & 1)) return 0.0f; b = 4 + __popc(y); }   // 5 + (pc(y)-1)
  return w[((size_t)n * 64 + i) * 9 + b];
}

__global__ __launch_bounds__(256) void ga_pack_weights(
    const float* __restrict__ wg1, const float* __restrict__ wg2,
    const float* __restrict__ wj1, const float* __restrict__ wj2,
    uint32_t* __restrict__ wpack)
{
  int idx = blockIdx.x * 256 + threadIdx.x;
  if (idx >= NPACK) return;
  int v     =  idx        & 7;
  int lane  = (idx >> 3)  & 31;
  int c     = (idx >> 8)  & 3;
  int nt    = (idx >> 10) & 3;
  int blade = (idx >> 12) & 15;
  int set   = (idx >> 16) & 3;
  const float* w = (set==0) ? wg1 : (set==1) ? wg2 : (set==2) ? wj1 : wj2;
  int n  = nt*16 + (lane & 15);
  int k0 = c*32 + ((lane & 16) ? 8 : 0) + (v < 4 ? v*2 : 16 + (v-4)*2);
  uint32_t lo = f2bf(wval(w, blade, k0,   n));
  uint32_t hi = f2bf(wval(w, blade, k0+1, n));
  wpack[idx] = lo | (hi << 16);
}

// ---------------------------------------------------------------------------
// A-fragment loader (16-bit A 16x32, ISA-documented layout):
// lane L: row m=L&15; elems 0..7 -> k = c*32 + (L>=16?8:0) + e,
//         elems 8..15 -> k = c*32 + 16 + (L>=16?8:0) + (e-8)
// k<64 reads blade y channels, k>=64 reads blade y^1 (the e0-partner rows).
// ---------------------------------------------------------------------------
__device__ __forceinline__ v16bf load_a_frag(const uint16_t* __restrict__ xs,
                                             int tokBase, int lane, int blade, int c){
  int hi8 = (lane & 16) ? 8 : 0;
  int kA  = c*32 + hi8;
  int kB  = kA + 16;
  int blA = (kA < 64) ? blade : (blade ^ 1);
  int blB = (kB < 64) ? blade : (blade ^ 1);
  FragB16 f;
  f.q[0] = *(const uint4*)(xs + tokBase + blA*64 + (kA & 63));
  f.q[1] = *(const uint4*)(xs + tokBase + blB*64 + (kB & 63));
  return f.v;
}

__device__ __forceinline__ v16bf load_b_frag(const uint32_t* __restrict__ wpack,
                                             int setG, int blade, int nt, int c, int lane){
  const uint4* p = (const uint4*)(wpack +
      (size_t)((((setG*16 + blade)*4 + nt)*4 + c)*32 + lane) * 8);
  FragB16 f;
  f.q[0] = p[0];
  f.q[1] = p[1];
  return f.v;
}

// ---------------------------------------------------------------------------
// Main kernel: one workgroup (8 wave32) per 16-token tile.
// ---------------------------------------------------------------------------
__global__ __launch_bounds__(256) void ga_bilinear_main(
    const float* __restrict__ x, const float* __restrict__ ref,
    const uint32_t* __restrict__ wpack, float* __restrict__ out)
{
  __shared__ __align__(16) uint16_t gbuf[2 * GB_SET];        // 98,304 B (bf16 transpose buffer)
  __shared__ __align__(16) uint16_t xs[TOK_TILE * XS_STRIDE]; // 33,024 B (bf16 x tile, [tok][blade][ch])

  const int tid  = threadIdx.x;
  const int lane = tid & 31;
  const int wave = tid >> 5;
  const int tok0 = blockIdx.x * TOK_TILE;

  // ---- stage x tile: [tok][ch][blade] f32 -> LDS [tok][blade][ch] bf16 ----
  // work item = (tok, chpair): read 2x16 contiguous floats, emit 16 packed dwords.
  uint32_t* xw = (uint32_t*)xs;
  for (int it = tid; it < TOK_TILE * 32; it += 256) {
    int tok = it >> 5, ch = (it & 31) * 2;
    const float4* px = (const float4*)(x + (((size_t)(tok0 + tok)) * 64 + ch) * 16);
    float a0[16], a1[16];
#pragma unroll
    for (int q = 0; q < 4; ++q) {
      *(float4*)(a0 + 4*q) = px[q];
      *(float4*)(a1 + 4*q) = px[q + 4];
    }
#pragma unroll
    for (int bl = 0; bl < 16; ++bl) {
      uint32_t d = (uint32_t)f2bf(a0[bl]) | ((uint32_t)f2bf(a1[bl]) << 16);
      xw[(tok * XS_STRIDE + bl * 64 + ch) >> 1] = d;
    }
  }
  __syncthreads();

  const int tokBase = (lane & 15) * XS_STRIDE;

  for (int phase = 0; phase < 2; ++phase) {
    // ---- equivariant linears: 128 jobs = 2 sets x 16 blades x 4 n-tiles ----
#pragma unroll 2
    for (int t = 0; t < 16; ++t) {
      int job   = wave * 16 + t;
      int setL  = job >> 6;
      int rem   = job & 63;
      int blade = rem >> 2;
      int nt    = rem & 3;
      int setG  = phase * 2 + setL;

      v8f acc = {0.f, 0.f, 0.f, 0.f, 0.f, 0.f, 0.f, 0.f};
#pragma unroll
      for (int c = 0; c < 4; ++c) {
        v16bf A = load_a_frag(xs, tokBase, lane, blade, c);
        v16bf B = load_b_frag(wpack, setG, blade, nt, c, lane);
        acc = __builtin_amdgcn_wmma_f32_16x16x32_bf16(
            /*neg_a=*/false, A, /*neg_b=*/false, B,
            /*c_mod=*/(short)0, acc, /*reuse_a=*/false, /*reuse_b=*/false);
      }
      // scatter D [tok 16 x outch 16] into gbuf[setL][tok][ch][blade]
      int chn = nt * 16 + (lane & 15);
      int tb  = (lane & 16) ? 8 : 0;
      uint16_t* gset = gbuf + setL * GB_SET;
#pragma unroll
      for (int r = 0; r < 8; ++r)
        gset[((tb + r) * 64 + chn) * GB_BS + blade] = f2bf(acc[r]);
    }
    __syncthreads();

    // ---- per-(token,channel) sparse Clifford bilinear ----
    for (int it = tid; it < TOK_TILE * 64; it += 256) {
      int tok = it >> 6, ch = it & 63;
      const uint16_t* pa = gbuf + ((size_t)tok * 64 + ch) * GB_BS;
      const uint16_t* pb = pa + GB_SET;
      FragB16 fa, fb;
      fa.q[0] = ((const uint4*)pa)[0]; fa.q[1] = ((const uint4*)pa)[1];
      fb.q[0] = ((const uint4*)pb)[0]; fb.q[1] = ((const uint4*)pb)[1];
      float a[16], b[16];
#pragma unroll
      for (int e = 0; e < 16; ++e) { a[e] = bf2f(fa.h[e]); b[e] = bf2f(fb.h[e]); }

      float o[16];
#pragma unroll
      for (int e = 0; e < 16; ++e) o[e] = 0.0f;

      float scale;
      int och;
      if (phase == 0) {
#pragma unroll
        for (int q = 0; q < 192; ++q)
          o[GPT.t[q].k] += GPT.t[q].s * a[GPT.t[q].i] * b[GPT.t[q].j];
        scale = 1e-5f;
        och = ch;
      } else {
#pragma unroll
        for (int q = 0; q < 81; ++q)
          o[JNT.t[q].k] += JNT.t[q].s * a[JNT.t[q].i] * b[JNT.t[q].j];
        scale = ref[(size_t)(tok0 + tok) * 16 + 15];   // pseudoscalar component
        och = 64 + ch;
      }

      float4* po = (float4*)(out + ((size_t)(tok0 + tok) * 128 + och) * 16);
#pragma unroll
      for (int q = 0; q < 4; ++q) {
        float4 v;
        v.x = o[4*q + 0] * scale; v.y = o[4*q + 1] * scale;
        v.z = o[4*q + 2] * scale; v.w = o[4*q + 3] * scale;
        po[q] = v;
      }
    }
    __syncthreads();   // gbuf reused by next phase
  }
}

// ---------------------------------------------------------------------------
extern "C" void kernel_launch(void* const* d_in, const int* in_sizes, int n_in,
                              void* d_out, int out_size, void* d_ws, size_t ws_size,
                              hipStream_t stream)
{
  const float* x   = (const float*)d_in[0];
  const float* ref = (const float*)d_in[1];
  const float* wg1 = (const float*)d_in[2];
  const float* wg2 = (const float*)d_in[3];
  const float* wj1 = (const float*)d_in[4];
  const float* wj2 = (const float*)d_in[5];
  float* out = (float*)d_out;

  uint32_t* wpack = (uint32_t*)d_ws;            // 1 MB packed bf16 weights

  ga_pack_weights<<<NPACK / 256, 256, 0, stream>>>(wg1, wg2, wj1, wj2, wpack);

  int ntok   = in_sizes[0] / (64 * 16);         // B*N = 16384
  int ntiles = ntok / TOK_TILE;                 // 1024 workgroups
  ga_bilinear_main<<<ntiles, 256, 0, stream>>>(x, ref, wpack, out);
}
